// HybridViTGATBlock_57561151701330
// MI455X (gfx1250) — matrix operand order repaired and checked by reading, not compiled, not run on hardware
//
#include <hip/hip_runtime.h>

// ---------------------------------------------------------------------------
// Types for CDNA5 WMMA (wave32): v16bf A/B fragments, v8f accumulator
// ---------------------------------------------------------------------------
typedef __attribute__((ext_vector_type(16))) __bf16 v16bf;
typedef __attribute__((ext_vector_type(8)))  float  v8f;

union FragBF { v16bf v; unsigned short u[16]; uint4 q[2]; };
union FragF  { v8f  v; float f[8]; };

__device__ __forceinline__ unsigned short f2bf(float x) {
    unsigned u = __float_as_uint(x);
    unsigned r = u + 0x7FFFu + ((u >> 16) & 1u);   // round to nearest even
    return (unsigned short)(r >> 16);
}
__device__ __forceinline__ float bf2f(unsigned short h) {
    return __uint_as_float(((unsigned)h) << 16);
}

// monotone float <-> uint key for atomicMax over floats
__device__ __forceinline__ unsigned fkey(float f) {
    unsigned u = __float_as_uint(f);
    return (u & 0x80000000u) ? ~u : (u | 0x80000000u);
}
__device__ __forceinline__ float funkey(unsigned k) {
    unsigned u = (k & 0x80000000u) ? (k & 0x7FFFFFFFu) : ~k;
    return __uint_as_float(u);
}

// ---------------------------------------------------------------------------
// f32 -> bf16 convert (weights)
// ---------------------------------------------------------------------------
__global__ void k_f32_to_bf16(const float* __restrict__ in,
                              unsigned short* __restrict__ out, int n) {
    int i = blockIdx.x * blockDim.x + threadIdx.x;
    if (i < n) out[i] = f2bf(in[i]);
}

__global__ void k_fill_u32(unsigned* p, unsigned v, int n) {
    int i = blockIdx.x * blockDim.x + threadIdx.x;
    if (i < n) p[i] = v;
}
__global__ void k_fill_f32(float* p, float v, int n) {
    int i = blockIdx.x * blockDim.x + threadIdx.x;
    if (i < n) p[i] = v;
}

// ---------------------------------------------------------------------------
// WMMA GEMM:  out[M,N] = A_bf16[M,K] @ B_bf16[K,N] (+bias) (+resid) (+act)
// block = 256 threads (8 waves); block tile 64x64; wave tile 16x32
// (two chained WMMAs per K-step sharing one A fragment); K step 32.
// act: 0 none, 1 ELU, 2 GELU(exact)
// ---------------------------------------------------------------------------
__global__ __launch_bounds__(256)
void k_gemm_bf16(const unsigned short* __restrict__ A,
                 const unsigned short* __restrict__ Bw,
                 const float* __restrict__ bias,
                 const float* __restrict__ resid,
                 float* __restrict__ outF,
                 unsigned short* __restrict__ outBF,
                 int M, int N, int K, int act)
{
    __shared__ unsigned short sA[64][40];   // padded: row stride 80B (16B mult)
    __shared__ unsigned short sBt[64][40];  // B tile stored transposed [n][k]

    const int blockM = blockIdx.y * 64;
    const int blockN = blockIdx.x * 64;
    const int tid  = threadIdx.x;
    const int wave = tid >> 5;
    const int lane = tid & 31;
    const int lm = lane & 15, lh = lane >> 4;
    const int wm = wave >> 1, wn = wave & 1;   // 4 x 2 wave grid

    FragF acc0, acc1;
#pragma unroll
    for (int i = 0; i < 8; ++i) { acc0.f[i] = 0.f; acc1.f[i] = 0.f; }

    const int arow = tid >> 2, akq = tid & 3;       // A: 64 rows x 4 chunks of 8
    const int bkk  = tid >> 3, bn0 = (tid & 7) * 8; // B: 32 k x 8 chunks of 8

    for (int k0 = 0; k0 < K; k0 += 32) {
        // stage A tile (row-major)
        uint4 av = *(const uint4*)(A + (size_t)(blockM + arow) * K + k0 + akq * 8);
        *(uint4*)&sA[arow][akq * 8] = av;
        // stage B tile transposed: 32(K) x 64(N) per step
        uint4 bv4 = *(const uint4*)(Bw + (size_t)(k0 + bkk) * N + blockN + bn0);
        unsigned short tb[8];
        *(uint4*)tb = bv4;
#pragma unroll
        for (int i = 0; i < 8; ++i) sBt[bn0 + i][bkk] = tb[i];

        if (k0 + 32 < K)   // emits global_prefetch_b8
            __builtin_prefetch((const void*)(A + (size_t)(blockM + arow) * K + k0 + 32), 0, 1);

        __syncthreads();

        FragBF a, b0, b1;
        a.q[0]  = *(const uint4*)&sA[wm * 16 + lm][lh * 8];
        a.q[1]  = *(const uint4*)&sA[wm * 16 + lm][16 + lh * 8];
        b0.q[0] = *(const uint4*)&sBt[wn * 32 + lm][lh * 8];
        b0.q[1] = *(const uint4*)&sBt[wn * 32 + lm][16 + lh * 8];
        b1.q[0] = *(const uint4*)&sBt[wn * 32 + 16 + lm][lh * 8];
        b1.q[1] = *(const uint4*)&sBt[wn * 32 + 16 + lm][16 + lh * 8];

        acc0.v = __builtin_amdgcn_wmma_f32_16x16x32_bf16(
            false, a.v, false, b0.v, (short)0, acc0.v, false, false);
        acc1.v = __builtin_amdgcn_wmma_f32_16x16x32_bf16(
            false, a.v, false, b1.v, (short)0, acc1.v, false, false);

        __syncthreads();
    }

#pragma unroll
    for (int half = 0; half < 2; ++half) {
        FragF* acc = half ? &acc1 : &acc0;
        const int gn = blockN + wn * 32 + half * 16 + lm;
#pragma unroll
        for (int r = 0; r < 8; ++r) {
            const int gm = blockM + wm * 16 + r + lh * 8;
            float v = acc->f[r];
            if (bias)  v += bias[gn];
            if (resid) v += resid[(size_t)gm * N + gn];
            if (act == 1)      v = v > 0.f ? v : (expf(v) - 1.f);
            else if (act == 2) v = 0.5f * v * (1.f + erff(v * 0.70710678118654752f));
            if (outF)  outF[(size_t)gm * N + gn] = v;
            if (outBF) outBF[(size_t)gm * N + gn] = f2bf(v);
        }
    }
}

// ---------------------------------------------------------------------------
// LayerNorm over last dim C for row-major [rows, C]; optional residual add,
// optional f32 + bf16 outputs.
// ---------------------------------------------------------------------------
__global__ __launch_bounds__(128)
void k_layernorm(const float* __restrict__ in, const float* __restrict__ resid,
                 const float* __restrict__ g, const float* __restrict__ bta,
                 float* __restrict__ outF, unsigned short* __restrict__ outBF, int C)
{
    const int row = blockIdx.x;
    const float* ip = in + (size_t)row * C;
    const float* rp = resid ? resid + (size_t)row * C : nullptr;
    __shared__ float s1[128], s2[128];
    float sum = 0.f, sq = 0.f;
    for (int c = threadIdx.x; c < C; c += 128) {
        float v = ip[c];
        if (rp) v += rp[c];
        sum += v; sq += v * v;
    }
    s1[threadIdx.x] = sum; s2[threadIdx.x] = sq;
    __syncthreads();
    for (int st = 64; st > 0; st >>= 1) {
        if (threadIdx.x < st) { s1[threadIdx.x] += s1[threadIdx.x + st];
                                s2[threadIdx.x] += s2[threadIdx.x + st]; }
        __syncthreads();
    }
    const float mu  = s1[0] / C;
    const float var = s2[0] / C - mu * mu;
    const float inv = rsqrtf(var + 1e-5f);
    for (int c = threadIdx.x; c < C; c += 128) {
        float v = ip[c];
        if (rp) v += rp[c];
        float y = (v - mu) * inv * g[c] + bta[c];
        if (outF)  outF[(size_t)row * C + c] = y;
        if (outBF) outBF[(size_t)row * C + c] = f2bf(y);
    }
}

// Input LN with NCHW->NSC permute: x[b][c][s] -> t_bf[b*S+s][c]
__global__ __launch_bounds__(128)
void k_ln_input(const float* __restrict__ x, const float* __restrict__ g,
                const float* __restrict__ bta, unsigned short* __restrict__ out,
                int C, int S)
{
    const int n = blockIdx.x;        // b*S + s
    const int bb = n / S, s = n - bb * S;
    const float* xp = x + (size_t)bb * C * S + s;
    __shared__ float s1[128], s2[128];
    float sum = 0.f, sq = 0.f;
    for (int c = threadIdx.x; c < C; c += 128) {
        float v = xp[(size_t)c * S];
        sum += v; sq += v * v;
    }
    s1[threadIdx.x] = sum; s2[threadIdx.x] = sq;
    __syncthreads();
    for (int st = 64; st > 0; st >>= 1) {
        if (threadIdx.x < st) { s1[threadIdx.x] += s1[threadIdx.x + st];
                                s2[threadIdx.x] += s2[threadIdx.x + st]; }
        __syncthreads();
    }
    const float mu  = s1[0] / C;
    const float var = s2[0] / C - mu * mu;
    const float inv = rsqrtf(var + 1e-5f);
    for (int c = threadIdx.x; c < C; c += 128) {
        float v = xp[(size_t)c * S];
        out[(size_t)n * C + c] = f2bf((v - mu) * inv * g[c] + bta[c]);
    }
}

// ---------------------------------------------------------------------------
// qkv split: qkv f32 [N,1536] -> q,k bf16 [B,H,S,64] (q scaled 1/8),
// vT bf16 [B,H,64,S]
// ---------------------------------------------------------------------------
__global__ void k_qkv_split(const float* __restrict__ qkv,
                            unsigned short* __restrict__ qo,
                            unsigned short* __restrict__ ko,
                            unsigned short* __restrict__ vto, int NN, int S)
{
    int i = blockIdx.x * blockDim.x + threadIdx.x;
    if (i >= NN * 512) return;
    int n = i >> 9, c = i & 511;
    int h = c >> 6, d = c & 63;
    int bb = n / S, s = n - bb * S;
    size_t bh = (size_t)bb * 8 + h;
    const float* qp = qkv + (size_t)n * 1536;
    qo[bh * S * 64 + (size_t)s * 64 + d]  = f2bf(qp[c] * 0.125f);
    ko[bh * S * 64 + (size_t)s * 64 + d]  = f2bf(qp[512 + c]);
    vto[bh * 64 * S + (size_t)d * S + s]  = f2bf(qp[1024 + c]);
}

// ---------------------------------------------------------------------------
// Streaming WMMA attention: one wave per (b, h, 16-query tile).
// scores = q@k^T (K=64 -> 2 chained wmma per 16-key subtile), online softmax,
// O += P@V via 4 wmma per 32-key step (V pre-transposed).
// ---------------------------------------------------------------------------
__global__ __launch_bounds__(32)
void k_attention(const unsigned short* __restrict__ qg,
                 const unsigned short* __restrict__ kg,
                 const unsigned short* __restrict__ vtg,
                 unsigned short* __restrict__ aoBF, int S)
{
    const int qt = blockIdx.x, h = blockIdx.y, bb = blockIdx.z;
    const size_t bh = (size_t)bb * 8 + h;
    const unsigned short* qp = qg  + bh * S * 64;
    const unsigned short* kp = kg  + bh * S * 64;
    const unsigned short* vp = vtg + bh * 64 * S;

    const int lane = threadIdx.x;
    const int lm = lane & 15, lh = lane >> 4;

    // q A-fragments (dh 0..31 and 32..63)
    FragBF aq0, aq1;
    {
        const int m = qt * 16 + lm;
        aq0.q[0] = *(const uint4*)(qp + (size_t)m * 64 + lh * 8);
        aq0.q[1] = *(const uint4*)(qp + (size_t)m * 64 + 16 + lh * 8);
        aq1.q[0] = *(const uint4*)(qp + (size_t)m * 64 + 32 + lh * 8);
        aq1.q[1] = *(const uint4*)(qp + (size_t)m * 64 + 48 + lh * 8);
    }

    FragF o[4];
#pragma unroll
    for (int t = 0; t < 4; ++t)
#pragma unroll
        for (int r = 0; r < 8; ++r) o[t].f[r] = 0.f;

    float mrow[8], lrow[8];
#pragma unroll
    for (int r = 0; r < 8; ++r) { mrow[r] = -1e30f; lrow[r] = 0.f; }

    __shared__ unsigned short sP[16][32];

    for (int kt = 0; kt < S; kt += 32) {
        FragF c0, c1;
#pragma unroll
        for (int j = 0; j < 2; ++j) {
            FragBF bk0, bk1;
            const int key = kt + j * 16 + lm;
            bk0.q[0] = *(const uint4*)(kp + (size_t)key * 64 + lh * 8);
            bk0.q[1] = *(const uint4*)(kp + (size_t)key * 64 + 16 + lh * 8);
            bk1.q[0] = *(const uint4*)(kp + (size_t)key * 64 + 32 + lh * 8);
            bk1.q[1] = *(const uint4*)(kp + (size_t)key * 64 + 48 + lh * 8);
            FragF cc;
#pragma unroll
            for (int r = 0; r < 8; ++r) cc.f[r] = 0.f;
            cc.v = __builtin_amdgcn_wmma_f32_16x16x32_bf16(
                false, aq1.v, false, bk1.v, (short)0, cc.v, false, false);
            cc.v = __builtin_amdgcn_wmma_f32_16x16x32_bf16(
                false, aq0.v, false, bk0.v, (short)0, cc.v, false, false);
            if (j == 0) c0 = cc; else c1 = cc;
        }

        // online softmax per row (rows r+8*lh live across the 16 lanes of a half)
#pragma unroll
        for (int r = 0; r < 8; ++r) {
            float s = fmaxf(c0.f[r], c1.f[r]);
#pragma unroll
            for (int off = 1; off < 16; off <<= 1)
                s = fmaxf(s, __shfl_xor(s, off, 32));
            const float mnew = fmaxf(mrow[r], s);
            const float scale = expf(mrow[r] - mnew);
            const float p0 = expf(c0.f[r] - mnew);
            const float p1 = expf(c1.f[r] - mnew);
            float rs = p0 + p1;
#pragma unroll
            for (int off = 1; off < 16; off <<= 1)
                rs += __shfl_xor(rs, off, 32);
            lrow[r] = lrow[r] * scale + rs;
            mrow[r] = mnew;
#pragma unroll
            for (int t = 0; t < 4; ++t) o[t].f[r] *= scale;
            sP[r + lh * 8][lm]      = f2bf(p0);
            sP[r + lh * 8][16 + lm] = f2bf(p1);
        }
        __syncthreads();

        FragBF ap;
        ap.q[0] = *(const uint4*)&sP[lm][lh * 8];
        ap.q[1] = *(const uint4*)&sP[lm][16 + lh * 8];
#pragma unroll
        for (int t = 0; t < 4; ++t) {
            FragBF bv;
            const int d = t * 16 + lm;
            bv.q[0] = *(const uint4*)(vp + (size_t)d * S + kt + lh * 8);
            bv.q[1] = *(const uint4*)(vp + (size_t)d * S + kt + 16 + lh * 8);
            o[t].v = __builtin_amdgcn_wmma_f32_16x16x32_bf16(
                false, ap.v, false, bv.v, (short)0, o[t].v, false, false);
        }
        __syncthreads();
    }

#pragma unroll
    for (int t = 0; t < 4; ++t)
#pragma unroll
        for (int r = 0; r < 8; ++r) {
            const int srow = qt * 16 + r + lh * 8;
            const float val = o[t].f[r] / lrow[r];
            aoBF[((size_t)bb * S + srow) * 512 + h * 64 + t * 16 + lm] = f2bf(val);
        }
}

// ---------------------------------------------------------------------------
// GAT helpers
// ---------------------------------------------------------------------------
__global__ void k_gat_satt(const float* __restrict__ hbuf,
                           const float* __restrict__ asrc,
                           const float* __restrict__ adst,
                           float* __restrict__ ssrc, float* __restrict__ sdst,
                           int NN, int heads)
{
    int i = blockIdx.x * blockDim.x + threadIdx.x;
    if (i >= NN * heads) return;
    int n = i / heads, h = i - n * heads;
    const float* hp = hbuf + ((size_t)n * heads + h) * 64;
    float a = 0.f, b = 0.f;
    for (int c = 0; c < 64; ++c) {
        a += hp[c] * asrc[h * 64 + c];
        b += hp[c] * adst[h * 64 + c];
    }
    ssrc[i] = a; sdst[i] = b;
}

__device__ __forceinline__ void edge_sd(const int* src, const int* dst, int e,
                                        int E, int& s, int& d) {
    if (e < E) { s = src[e]; d = dst[e]; } else { s = d = e - E; }
}

__global__ void k_gat_max(const int* __restrict__ src, const int* __restrict__ dst,
                          int E, int ET, int heads,
                          const float* __restrict__ ssrc, const float* __restrict__ sdst,
                          unsigned* __restrict__ maxb)
{
    int i = blockIdx.x * blockDim.x + threadIdx.x;
    if (i >= ET * heads) return;
    int e = i / heads, h = i - e * heads, s, d;
    edge_sd(src, dst, e, E, s, d);
    float v = ssrc[s * heads + h] + sdst[d * heads + h];
    v = v > 0.f ? v : 0.2f * v;
    atomicMax(&maxb[d * heads + h], fkey(v));
}

__global__ void k_gat_sum(const int* __restrict__ src, const int* __restrict__ dst,
                          int E, int ET, int heads,
                          const float* __restrict__ ssrc, const float* __restrict__ sdst,
                          const unsigned* __restrict__ maxb, float* __restrict__ den)
{
    int i = blockIdx.x * blockDim.x + threadIdx.x;
    if (i >= ET * heads) return;
    int e = i / heads, h = i - e * heads, s, d;
    edge_sd(src, dst, e, E, s, d);
    float v = ssrc[s * heads + h] + sdst[d * heads + h];
    v = v > 0.f ? v : 0.2f * v;
    atomicAdd(&den[d * heads + h], expf(v - funkey(maxb[d * heads + h])));
}

__global__ void k_gat_scatter(const int* __restrict__ src, const int* __restrict__ dst,
                              int E, int ET, int heads,
                              const float* __restrict__ ssrc, const float* __restrict__ sdst,
                              const unsigned* __restrict__ maxb, const float* __restrict__ den,
                              const float* __restrict__ hbuf, float* __restrict__ agg)
{
    int i = blockIdx.x * blockDim.x + threadIdx.x;
    if (i >= ET * heads) return;
    int e = i / heads, h = i - e * heads, s, d;
    edge_sd(src, dst, e, E, s, d);
    float v = ssrc[s * heads + h] + sdst[d * heads + h];
    v = v > 0.f ? v : 0.2f * v;
    const float w = expf(v - funkey(maxb[d * heads + h]));
    const float alpha = w / (den[d * heads + h] + 1e-16f);
    const float* hp = hbuf + ((size_t)s * heads + h) * 64;
    float* op = agg + ((size_t)d * heads + h) * 64;
    for (int c = 0; c < 64; ++c) atomicAdd(&op[c], hp[c] * alpha);
}

__global__ void k_bias_elu_bf(const float* __restrict__ in, const float* __restrict__ bias,
                              unsigned short* __restrict__ out, int total, int C)
{
    int i = blockIdx.x * blockDim.x + threadIdx.x;
    if (i >= total) return;
    float v = in[i] + bias[i % C];
    v = v > 0.f ? v : (expf(v) - 1.f);
    out[i] = f2bf(v);
}

__global__ void k_bias_add_f(const float* __restrict__ in, const float* __restrict__ bias,
                             float* __restrict__ out, int total, int C)
{
    int i = blockIdx.x * blockDim.x + threadIdx.x;
    if (i >= total) return;
    out[i] = in[i] + bias[i % C];
}

// final permute: y [B*S, 64] -> out [B, 64, H, W]
__global__ void k_out_perm(const float* __restrict__ y, float* __restrict__ out,
                           int Bb, int CG, int Hh, int Ww)
{
    int i = blockIdx.x * blockDim.x + threadIdx.x;
    int total = Bb * CG * Hh * Ww;
    if (i >= total) return;
    int w = i % Ww; int t = i / Ww;
    int hh = t % Hh; t /= Hh;
    int c = t % CG; int bb = t / CG;
    out[i] = y[((size_t)bb * Hh * Ww + hh * Ww + w) * CG + c];
}

// ---------------------------------------------------------------------------
// Host orchestration
// ---------------------------------------------------------------------------
extern "C" void kernel_launch(void* const* d_in, const int* in_sizes, int n_in,
                              void* d_out, int out_size, void* d_ws, size_t ws_size,
                              hipStream_t stream)
{
    (void)n_in; (void)out_size; (void)ws_size;
    const int Bb = 2, Hh = 48, Ww = 48, S = Hh * Ww, NN = Bb * S; // 4608

    // ---- inputs (setup_inputs dict order) ----
    const float* x      = (const float*)d_in[0];
    const float* n1_g   = (const float*)d_in[1];
    const float* n1_b   = (const float*)d_in[2];
    const float* proj_w = (const float*)d_in[3];
    const float* proj_b = (const float*)d_in[4];
    const float* qkv_w  = (const float*)d_in[5];
    const float* qkv_b  = (const float*)d_in[6];
    const float* out_w  = (const float*)d_in[7];
    const float* out_b  = (const float*)d_in[8];
    const float* n2_g   = (const float*)d_in[9];
    const float* n2_b   = (const float*)d_in[10];
    const float* gpl_w  = (const float*)d_in[11];
    const float* gpl_b  = (const float*)d_in[12];
    const float* g1_w   = (const float*)d_in[13];
    const float* g1_as  = (const float*)d_in[14];
    const float* g1_ad  = (const float*)d_in[15];
    const float* g1_b   = (const float*)d_in[16];
    const float* pg_w   = (const float*)d_in[17];
    const float* pg_b   = (const float*)d_in[18];
    const float* g2_w   = (const float*)d_in[19];
    const float* g2_as  = (const float*)d_in[20];
    const float* g2_ad  = (const float*)d_in[21];
    const float* g2_b   = (const float*)d_in[22];
    const float* res_w  = (const float*)d_in[23];
    const float* res_b  = (const float*)d_in[24];
    const float* n3_g   = (const float*)d_in[25];
    const float* n3_b   = (const float*)d_in[26];
    const float* m1_w   = (const float*)d_in[27];
    const float* m1_b   = (const float*)d_in[28];
    const float* m2_w   = (const float*)d_in[29];
    const float* m2_b   = (const float*)d_in[30];
    const float* m3_w   = (const float*)d_in[31];
    const float* m3_b   = (const float*)d_in[32];
    const float* nf_g   = (const float*)d_in[33];
    const float* nf_b   = (const float*)d_in[34];
    const int*   eidx   = (const int*)d_in[35];
    const int E  = in_sizes[35] / 2;
    const int ET = E + NN;                 // + self loops
    const int* esrc = eidx;
    const int* edst = eidx + E;

    float* outp = (float*)d_out;

    // ---- workspace bump allocator ----
    char* base = (char*)d_ws;
    size_t off = 0;
    auto alloc = [&](size_t bytes) -> void* {
        off = (off + 255) & ~(size_t)255;
        void* p = base + off;
        off += bytes;
        return p;
    };
    auto abf = [&](size_t n) { return (unsigned short*)alloc(n * 2); };
    auto af  = [&](size_t n) { return (float*)alloc(n * 4); };
    auto au  = [&](size_t n) { return (unsigned*)alloc(n * 4); };

    unsigned short* t_bf   = abf((size_t)NN * 512);
    unsigned short* wproj  = abf(512 * 512);
    unsigned short* wqkv   = abf(512 * 1536);
    unsigned short* wout   = abf(512 * 512);
    unsigned short* wgpl   = abf(512 * 512);
    unsigned short* wg1    = abf(512 * 512);
    unsigned short* wpg    = abf(512 * 512);
    unsigned short* wg2    = abf(512 * 64);
    unsigned short* wres   = abf(512 * 64);
    unsigned short* wm1    = abf(64 * 64);
    unsigned short* wm2    = abf(64 * 128);
    unsigned short* wm3    = abf(128 * 64);
    float*          xv_f   = af((size_t)NN * 512);
    unsigned short* xv_bf  = abf((size_t)NN * 512);
    float*          qkv_f  = af((size_t)NN * 1536);
    unsigned short* q_bf   = abf((size_t)NN * 512);
    unsigned short* k_bf   = abf((size_t)NN * 512);
    unsigned short* vT_bf  = abf((size_t)NN * 512);
    unsigned short* ao_bf  = abf((size_t)NN * 512);
    float*          xres_f = af((size_t)NN * 512);    // nf
    unsigned short* nf_bf  = abf((size_t)NN * 512);
    unsigned short* gln_bf = abf((size_t)NN * 512);
    unsigned short* g_bf   = abf((size_t)NN * 512);
    float*          h1_f   = af((size_t)NN * 512);
    float*          ssrc1  = af((size_t)NN * 8);
    float*          sdst1  = af((size_t)NN * 8);
    unsigned*       maxb1  = au((size_t)NN * 8);
    float*          den1   = af((size_t)NN * 8);
    float*          agg1   = af((size_t)NN * 512);
    unsigned short* g1_bf  = abf((size_t)NN * 512);
    unsigned short* g2in_bf= abf((size_t)NN * 512);
    float*          h2_f   = af((size_t)NN * 64);
    float*          ssrc2  = af(NN);
    float*          sdst2  = af(NN);
    unsigned*       maxb2  = au(NN);
    float*          den2   = af(NN);
    float*          agg2   = af((size_t)NN * 64);
    float*          g2res  = af((size_t)NN * 64);
    float*          ypre   = af((size_t)NN * 64);
    float*          y_f    = af((size_t)NN * 64);
    unsigned short* y_bf   = abf((size_t)NN * 64);
    unsigned short* m1_bf  = abf((size_t)NN * 64);
    unsigned short* m2_bf  = abf((size_t)NN * 128);
    float*          m3_f   = af((size_t)NN * 64);
    float*          yfin   = af((size_t)NN * 64);

    auto grid1 = [](int n) { return dim3((unsigned)((n + 255) / 256)); };
    auto conv = [&](const float* w, unsigned short* o, int n) {
        k_f32_to_bf16<<<grid1(n), 256, 0, stream>>>(w, o, n);
    };
    auto gemm = [&](const unsigned short* A, const unsigned short* Bw,
                    const float* bias, const float* resid,
                    float* oF, unsigned short* oBF, int M, int Nc, int K, int act) {
        dim3 g(Nc / 64, M / 64);
        k_gemm_bf16<<<g, 256, 0, stream>>>(A, Bw, bias, resid, oF, oBF, M, Nc, K, act);
    };

    // 1) weight conversion to bf16
    conv(proj_w, wproj, 512 * 512);
    conv(qkv_w,  wqkv,  512 * 1536);
    conv(out_w,  wout,  512 * 512);
    conv(gpl_w,  wgpl,  512 * 512);
    conv(g1_w,   wg1,   512 * 512);
    conv(pg_w,   wpg,   512 * 512);
    conv(g2_w,   wg2,   512 * 64);
    conv(res_w,  wres,  512 * 64);
    conv(m1_w,   wm1,   64 * 64);
    conv(m2_w,   wm2,   64 * 128);
    conv(m3_w,   wm3,   128 * 64);

    // 2) LN1 over permuted input
    k_ln_input<<<NN, 128, 0, stream>>>(x, n1_g, n1_b, t_bf, 512, S);

    // 3) proj: xv = t @ proj_w + proj_b
    gemm(t_bf, wproj, proj_b, nullptr, xv_f, xv_bf, NN, 512, 512, 0);

    // 4) qkv = xv @ qkv_w + qkv_b
    gemm(xv_bf, wqkv, qkv_b, nullptr, qkv_f, nullptr, NN, 1536, 512, 0);

    // 5) split + pre-scale + V transpose
    k_qkv_split<<<grid1(NN * 512), 256, 0, stream>>>(qkv_f, q_bf, k_bf, vT_bf, NN, S);

    // 6) streaming WMMA attention
    {
        dim3 g(S / 16, 8, Bb);
        k_attention<<<g, 32, 0, stream>>>(q_bf, k_bf, vT_bf, ao_bf, S);
    }

    // 7) out proj + residual: xres = xv + ao @ out_w + out_b   (nf)
    gemm(ao_bf, wout, out_b, xv_f, xres_f, nf_bf, NN, 512, 512, 0);

    // 8) LN2(nf) -> bf16
    k_layernorm<<<NN, 128, 0, stream>>>(xres_f, nullptr, n2_g, n2_b, nullptr, gln_bf, 512);

    // 9) g = LN2(nf) @ gpl_w + gpl_b
    gemm(gln_bf, wgpl, gpl_b, nullptr, nullptr, g_bf, NN, 512, 512, 0);

    // 10) GAT1 linear: h1 = g @ g1_w
    gemm(g_bf, wg1, nullptr, nullptr, h1_f, nullptr, NN, 512, 512, 0);

    // 11) GAT1 segment softmax + scatter
    k_gat_satt<<<grid1(NN * 8), 256, 0, stream>>>(h1_f, g1_as, g1_ad, ssrc1, sdst1, NN, 8);
    k_fill_u32<<<grid1(NN * 8), 256, 0, stream>>>(maxb1, 0u, NN * 8);
    k_fill_f32<<<grid1(NN * 8), 256, 0, stream>>>(den1, 0.f, NN * 8);
    k_fill_f32<<<grid1(NN * 512), 256, 0, stream>>>(agg1, 0.f, NN * 512);
    k_gat_max    <<<grid1(ET * 8), 256, 0, stream>>>(esrc, edst, E, ET, 8, ssrc1, sdst1, maxb1);
    k_gat_sum    <<<grid1(ET * 8), 256, 0, stream>>>(esrc, edst, E, ET, 8, ssrc1, sdst1, maxb1, den1);
    k_gat_scatter<<<grid1(ET * 8), 256, 0, stream>>>(esrc, edst, E, ET, 8, ssrc1, sdst1, maxb1, den1, h1_f, agg1);
    k_bias_elu_bf<<<grid1(NN * 512), 256, 0, stream>>>(agg1, g1_b, g1_bf, NN * 512, 512);

    // 12) pg: g2in = g1 @ pg_w + pg_b
    gemm(g1_bf, wpg, pg_b, nullptr, nullptr, g2in_bf, NN, 512, 512, 0);

    // 13) GAT2 linear: h2 = g2in @ g2_w   (1 head, 64 ch)
    gemm(g2in_bf, wg2, nullptr, nullptr, h2_f, nullptr, NN, 64, 512, 0);

    // 14) GAT2 segment softmax + scatter (heads=1)
    k_gat_satt<<<grid1(NN), 256, 0, stream>>>(h2_f, g2_as, g2_ad, ssrc2, sdst2, NN, 1);
    k_fill_u32<<<grid1(NN), 256, 0, stream>>>(maxb2, 0u, NN);
    k_fill_f32<<<grid1(NN), 256, 0, stream>>>(den2, 0.f, NN);
    k_fill_f32<<<grid1(NN * 64), 256, 0, stream>>>(agg2, 0.f, NN * 64);
    k_gat_max    <<<grid1(ET), 256, 0, stream>>>(esrc, edst, E, ET, 1, ssrc2, sdst2, maxb2);
    k_gat_sum    <<<grid1(ET), 256, 0, stream>>>(esrc, edst, E, ET, 1, ssrc2, sdst2, maxb2, den2);
    k_gat_scatter<<<grid1(ET), 256, 0, stream>>>(esrc, edst, E, ET, 1, ssrc2, sdst2, maxb2, den2, h2_f, agg2);
    k_bias_add_f<<<grid1(NN * 64), 256, 0, stream>>>(agg2, g2_b, g2res, NN * 64, 64);

    // 15) ypre = nf @ res_w + res_b + g2 ; y = LN3(ypre)
    gemm(nf_bf, wres, res_b, g2res, ypre, nullptr, NN, 64, 512, 0);
    k_layernorm<<<NN, 128, 0, stream>>>(ypre, nullptr, n3_g, n3_b, y_f, y_bf, 64);

    // 16) MLP: gelu, gelu, linear
    gemm(y_bf, wm1, m1_b, nullptr, nullptr, m1_bf, NN, 64, 64, 2);
    gemm(m1_bf, wm2, m2_b, nullptr, nullptr, m2_bf, NN, 128, 64, 2);
    gemm(m2_bf, wm3, m3_b, nullptr, m3_f, nullptr, NN, 64, 128, 0);

    // 17) final LN(y + m) and output permute [B,S,64] -> [B,64,H,W]
    k_layernorm<<<NN, 128, 0, stream>>>(y_f, m3_f, nf_g, nf_b, yfin, nullptr, 64);
    k_out_perm<<<grid1(Bb * 64 * Hh * Ww), 256, 0, stream>>>(yfin, outp, Bb, 64, Hh, Ww);
}
